// TransformerAllocator_46634754900169
// MI455X (gfx1250) — matrix operand order repaired and checked by reading, not compiled, not run on hardware
//
#include <hip/hip_runtime.h>
#include <hip/hip_bf16.h>

// ---------------- model constants ----------------
constexpr int BB  = 8;
constexpr int TT  = 1024;
constexpr int FF  = 128;
constexpr int DD  = 512;
constexpr int HH  = 8;
constexpr int LL  = 2;
constexpr int DFFC = 2048;
constexpr int AA  = 16;
constexpr int HDC = DD / HH;       // 64
constexpr int MM  = BB * TT;       // 8192 rows
constexpr float LN_EPS = 1e-5f;

// ---------------- WMMA types ----------------
typedef __attribute__((ext_vector_type(16))) _Float16 v16h;
typedef __attribute__((ext_vector_type(8)))  float    v8f;
typedef __attribute__((ext_vector_type(4)))  unsigned int u32x4;

union FragU { v16h v; u32x4 q[2]; };

__device__ inline v8f wmma_f16(v16h a, v16h b, v8f c) {
  // 8 args: (neg_a, A, neg_b, B, c_mod, C, reuse_a, reuse_b)
  return __builtin_amdgcn_wmma_f32_16x16x32_f16(false, a, false, b, (short)0, c,
                                                false, false);
}

// A operand (16x32, MxK), source row-major M x K, base -> element (m0,k0).
// Layout: lane<16: h0-15 -> K0-15; lane>=16: h0-15 -> K16-31.
__device__ inline v16h load_afrag(const _Float16* __restrict__ base, int stride,
                                  int lane) {
  int r = lane & 15, g = lane >> 4;
  const _Float16* p = base + (size_t)r * stride + 16 * g;
  FragU f;
  f.q[0] = *(const u32x4*)(p);
  f.q[1] = *(const u32x4*)(p + 8);
  return f.v;
}

// B operand (32x16, KxN) from storage that is row-major N x K (e.g. W[N][K]).
// Layout (contiguous k, same as load_afrag): lane<16: col n=lane, K=0..15; lane>=16: K=16..31.
__device__ inline v16h load_bfrag_nk(const _Float16* __restrict__ base, int stride,
                                     int lane) {
  int n = lane & 15, g = lane >> 4;
  const _Float16* p = base + (size_t)n * stride + 16 * g;
  FragU f;
  f.q[0] = *(const u32x4*)(p);
  f.q[1] = *(const u32x4*)(p + 8);
  return f.v;
}

// B operand (32x16, KxN) from storage that is row-major K x N (e.g. V[k][d]).
__device__ inline v16h load_bfrag_kn(const _Float16* __restrict__ base, int stride,
                                     int lane) {
  int n = lane & 15, g = lane >> 4;
  FragU f;
#pragma unroll
  for (int hh = 0; hh < 16; ++hh) {
    int k = hh + 16 * g;
    f.v[hh] = base[(size_t)k * stride + n];
  }
  return f.v;
}

// ---------------- f32 -> f16 convert ----------------
__global__ void f32_to_f16_kernel(const float* __restrict__ src,
                                  _Float16* __restrict__ dst, int n) {
  int i = blockIdx.x * blockDim.x + threadIdx.x;
  if (i < n) dst[i] = (_Float16)src[i];
}

// ---------------- WMMA GEMM: C[M,N] = A[M,K] @ W[N,K]^T + bias ----------------
// 8 waves / block; each wave owns a 32(m) x 64(n) tile:
//   2 A-fragments x 4 B-fragments -> 8 WMMA per 32-K step
//   (B reused 2x, A reused 4x in registers -> ~19 flop/byte of L2 traffic)
template <bool RELU, bool OUTF, bool OUT16>
__global__ __launch_bounds__(256) void gemm_wmma(
    const _Float16* __restrict__ A, const _Float16* __restrict__ W,
    const float* __restrict__ bias, float* __restrict__ Cf,
    _Float16* __restrict__ C16, int Mm, int Nn, int Kk) {
  const int lane = threadIdx.x & 31;
  const int wave = threadIdx.x >> 5;
  const int n0 = blockIdx.x * 64;
  const int m0 = (blockIdx.y * 8 + wave) * 32;
  if (m0 >= Mm) return;

  v8f acc[2][4];
#pragma unroll
  for (int mt = 0; mt < 2; ++mt)
#pragma unroll
    for (int nt = 0; nt < 4; ++nt) { v8f z = {}; acc[mt][nt] = z; }

  const _Float16* Arow = A + (size_t)m0 * Kk;
  for (int k0 = 0; k0 < Kk; k0 += 32) {
    __builtin_prefetch(Arow + k0 + 64, 0, 0);
    v16h af[2];
#pragma unroll
    for (int mt = 0; mt < 2; ++mt)
      af[mt] = load_afrag(Arow + (size_t)mt * 16 * Kk + k0, Kk, lane);
#pragma unroll
    for (int nt = 0; nt < 4; ++nt) {
      v16h bf = load_bfrag_nk(W + (size_t)(n0 + nt * 16) * Kk + k0, Kk, lane);
#pragma unroll
      for (int mt = 0; mt < 2; ++mt)
        acc[mt][nt] = wmma_f16(af[mt], bf, acc[mt][nt]);
    }
  }

  const int nl = lane & 15, g = lane >> 4;
#pragma unroll
  for (int mt = 0; mt < 2; ++mt) {
#pragma unroll
    for (int nt = 0; nt < 4; ++nt) {
      int n = n0 + nt * 16 + nl;
      float bv = bias ? bias[n] : 0.f;
#pragma unroll
      for (int r = 0; r < 8; ++r) {
        float v = acc[mt][nt][r] + bv;
        if (RELU) v = fmaxf(v, 0.f);
        size_t idx = (size_t)(m0 + mt * 16 + r + 8 * g) * Nn + n;
        if (OUTF)  Cf[idx]  = v;
        if (OUT16) C16[idx] = (_Float16)v;
      }
    }
  }
}

// ---------------- positional encoding + f16 mirror ----------------
__global__ void add_pe_kernel(float* __restrict__ hf, _Float16* __restrict__ h16) {
  int i = blockIdx.x * blockDim.x + threadIdx.x;
  if (i >= MM * DD) return;
  int d = i % DD;
  int t = (i / DD) % TT;
  int j = d & ~1;
  float ang = (float)t * __expf(-(float)j * (9.2103403719761836f / (float)DD));
  float pe = (d & 1) ? __cosf(ang) : __sinf(ang);
  float v = hf[i] + pe;
  hf[i] = v;
  h16[i] = (_Float16)v;
}

// ---------------- LayerNorm(residual) ----------------
__global__ __launch_bounds__(256) void ln_residual_kernel(
    float* __restrict__ hf, const float* __restrict__ add,
    const float* __restrict__ gam, const float* __restrict__ bta,
    _Float16* __restrict__ h16) {
  const int row = blockIdx.x;
  const int tid = threadIdx.x;
  float* x = hf + (size_t)row * DD;
  const float* a2 = add + (size_t)row * DD;
  float v[2];
  float s = 0.f, ss = 0.f;
#pragma unroll
  for (int i = 0; i < 2; ++i) {
    int d = tid + i * 256;
    float t = x[d] + a2[d];
    v[i] = t;
    s += t; ss += t * t;
  }
  __shared__ float red[16];
#pragma unroll
  for (int m = 1; m < 32; m <<= 1) { s += __shfl_xor(s, m); ss += __shfl_xor(ss, m); }
  int wv = tid >> 5;
  if ((tid & 31) == 0) { red[wv] = s; red[8 + wv] = ss; }
  __syncthreads();
  if (tid < 32) {
    float ps  = (tid < 8) ? red[tid] : 0.f;
    float pss = (tid < 8) ? red[8 + tid] : 0.f;
#pragma unroll
    for (int m = 1; m < 8; m <<= 1) { ps += __shfl_xor(ps, m); pss += __shfl_xor(pss, m); }
    if (tid == 0) { red[0] = ps; red[1] = pss; }
  }
  __syncthreads();
  float mu = red[0] / (float)DD;
  float var = red[1] / (float)DD - mu * mu;
  float rstd = rsqrtf(var + LN_EPS);
#pragma unroll
  for (int i = 0; i < 2; ++i) {
    int d = tid + i * 256;
    float y = (v[i] - mu) * rstd * gam[d] + bta[d];
    x[d] = y;
    h16[(size_t)row * DD + d] = (_Float16)y;
  }
}

// ---------------- flash attention (WMMA, online softmax) ----------------
__global__ __launch_bounds__(128) void flash_attn_kernel(
    const _Float16* __restrict__ qkv, _Float16* __restrict__ out) {
  const int lane = threadIdx.x & 31;
  const int wave = threadIdx.x >> 5;
  const int idx = blockIdx.x;          // B*H*(T/64) blocks
  const int qb = idx & 15;
  const int h  = (idx >> 4) & 7;
  const int b  = idx >> 7;
  const int q0 = qb * 64 + wave * 16;
  const int st = 3 * DD;               // 1536 row stride in qkv
  const _Float16* base = qkv + (size_t)b * TT * st;
  const _Float16* Qb = base + h * HDC;
  const _Float16* Kb = base + DD + h * HDC;
  const _Float16* Vb = base + 2 * DD + h * HDC;

  __shared__ alignas(16) _Float16 pst[4][16 * 32];
  _Float16* myp = pst[wave];

  v16h qf0 = load_afrag(Qb + (size_t)q0 * st, st, lane);
  v16h qf1 = load_afrag(Qb + (size_t)q0 * st + 32, st, lane);

  float mrun[8], lrun[8];
#pragma unroll
  for (int r = 0; r < 8; ++r) { mrun[r] = -1e30f; lrun[r] = 0.f; }
  v8f o[4];
#pragma unroll
  for (int i = 0; i < 4; ++i) { v8f z = {}; o[i] = z; }

  const float scale = 0.125f;          // 1/sqrt(64)
  const int nl = lane & 15, g = lane >> 4;

  for (int kk = 0; kk < TT; kk += 32) {
    v8f s0 = {}, s1 = {};
    v16h kf;
    kf = load_bfrag_nk(Kb + (size_t)kk * st, st, lane);
    s0 = wmma_f16(qf0, kf, s0);
    kf = load_bfrag_nk(Kb + (size_t)kk * st + 32, st, lane);
    s0 = wmma_f16(qf1, kf, s0);
    kf = load_bfrag_nk(Kb + (size_t)(kk + 16) * st, st, lane);
    s1 = wmma_f16(qf0, kf, s1);
    kf = load_bfrag_nk(Kb + (size_t)(kk + 16) * st + 32, st, lane);
    s1 = wmma_f16(qf1, kf, s1);

#pragma unroll
    for (int r = 0; r < 8; ++r) {
      float a0 = s0[r] * scale, a1 = s1[r] * scale;
      float tm = fmaxf(a0, a1);
      tm = fmaxf(tm, __shfl_xor(tm, 1));
      tm = fmaxf(tm, __shfl_xor(tm, 2));
      tm = fmaxf(tm, __shfl_xor(tm, 4));
      tm = fmaxf(tm, __shfl_xor(tm, 8));
      float mnew  = fmaxf(mrun[r], tm);
      float alpha = __expf(mrun[r] - mnew);
      float p0 = __expf(a0 - mnew);
      float p1 = __expf(a1 - mnew);
      float rs = p0 + p1;
      rs += __shfl_xor(rs, 1);
      rs += __shfl_xor(rs, 2);
      rs += __shfl_xor(rs, 4);
      rs += __shfl_xor(rs, 8);
      lrun[r] = lrun[r] * alpha + rs;
      mrun[r] = mnew;
#pragma unroll
      for (int nt = 0; nt < 4; ++nt) o[nt][r] *= alpha;
      int m = r + 8 * g;               // C-layout row owned by this lane
      myp[m * 32 + nl]      = (_Float16)p0;
      myp[m * 32 + 16 + nl] = (_Float16)p1;
    }
    asm volatile("s_wait_dscnt 0x0" ::: "memory");   // CDNA5 split DS counter

    v16h pf;                            // P in A-format from LDS
    {
      const _Float16* pp = myp + nl * 32 + 16 * g;
      FragU f;
      f.q[0] = *(const u32x4*)(pp);
      f.q[1] = *(const u32x4*)(pp + 8);
      pf = f.v;
    }
#pragma unroll
    for (int nt = 0; nt < 4; ++nt) {
      v16h vf = load_bfrag_kn(Vb + (size_t)kk * st + nt * 16, st, lane);
      o[nt] = wmma_f16(pf, vf, o[nt]);
    }
  }

  _Float16* ob = out + (size_t)b * TT * DD + h * HDC;
#pragma unroll
  for (int nt = 0; nt < 4; ++nt) {
#pragma unroll
    for (int r = 0; r < 8; ++r) {
      float v = o[nt][r] / lrun[r];
      ob[(size_t)(q0 + r + 8 * g) * DD + nt * 16 + nl] = (_Float16)v;
    }
  }
}

// ---------------- head: hid = relu(last @ mlp1^T + b1) ----------------
__global__ __launch_bounds__(256) void head1_kernel(
    const float* __restrict__ hf, const float* __restrict__ w1,
    const float* __restrict__ b1, float* __restrict__ hid) {
  int b = blockIdx.x;
  const float* xr = hf + ((size_t)b * TT + (TT - 1)) * DD;
  for (int j = threadIdx.x; j < DD; j += 256) {
    const float* wr = w1 + (size_t)j * DD;
    float acc = b1[j];
    for (int k = 0; k < DD; ++k) acc += xr[k] * wr[k];
    hid[b * DD + j] = fmaxf(acc, 0.f);
  }
}

// ---------------- head: logits + softmax ----------------
__global__ __launch_bounds__(128) void head2_kernel(
    const float* __restrict__ hid, const float* __restrict__ w2,
    const float* __restrict__ b2, float* __restrict__ outp) {
  int tid = threadIdx.x;
  int b = tid >> 4;            // 0..7
  int a = tid & 15;            // 0..15
  const float* xr = hid + b * DD;
  const float* wr = w2 + a * DD;
  float acc = b2[a];
  for (int k = 0; k < DD; ++k) acc += xr[k] * wr[k];
  float mx = acc;
  mx = fmaxf(mx, __shfl_xor(mx, 1));
  mx = fmaxf(mx, __shfl_xor(mx, 2));
  mx = fmaxf(mx, __shfl_xor(mx, 4));
  mx = fmaxf(mx, __shfl_xor(mx, 8));
  float e = __expf(acc - mx);
  float sm = e;
  sm += __shfl_xor(sm, 1);
  sm += __shfl_xor(sm, 2);
  sm += __shfl_xor(sm, 4);
  sm += __shfl_xor(sm, 8);
  outp[b * AA + a] = e / sm;
}

// ---------------- host-side orchestration ----------------
extern "C" void kernel_launch(void* const* d_in, const int* in_sizes, int n_in,
                              void* d_out, int out_size, void* d_ws, size_t ws_size,
                              hipStream_t stream) {
  (void)in_sizes; (void)n_in; (void)out_size; (void)ws_size;
  const float* x          = (const float*)d_in[0];
  const float* proj_w     = (const float*)d_in[1];
  const float* proj_b     = (const float*)d_in[2];
  const float* in_proj_w  = (const float*)d_in[3];
  const float* in_proj_b  = (const float*)d_in[4];
  const float* out_proj_w = (const float*)d_in[5];
  const float* out_proj_b = (const float*)d_in[6];
  const float* ln1_w = (const float*)d_in[7];
  const float* ln1_b = (const float*)d_in[8];
  const float* ln2_w = (const float*)d_in[9];
  const float* ln2_b = (const float*)d_in[10];
  const float* ff1_w = (const float*)d_in[11];
  const float* ff1_b = (const float*)d_in[12];
  const float* ff2_w = (const float*)d_in[13];
  const float* ff2_b = (const float*)d_in[14];
  const float* mlp1_w = (const float*)d_in[15];
  const float* mlp1_b = (const float*)d_in[16];
  const float* mlp2_w = (const float*)d_in[17];
  const float* mlp2_b = (const float*)d_in[18];
  float* outp = (float*)d_out;

  char* wsb = (char*)d_ws;
  size_t off = 0;
  auto take = [&](size_t nbytes) -> char* {
    char* p = wsb + off;
    off += (nbytes + 255) & ~(size_t)255;
    return p;
  };
  _Float16* x16    = (_Float16*)take((size_t)MM * FF * 2);
  _Float16* wp16   = (_Float16*)take((size_t)DD * FF * 2);
  _Float16* wqkv16 = (_Float16*)take((size_t)LL * 3 * DD * DD * 2);
  _Float16* wo16   = (_Float16*)take((size_t)LL * DD * DD * 2);
  _Float16* wf1_16 = (_Float16*)take((size_t)LL * DFFC * DD * 2);
  _Float16* wf2_16 = (_Float16*)take((size_t)LL * DD * DFFC * 2);
  float*    hf     = (float*)take((size_t)MM * DD * 4);
  _Float16* h16    = (_Float16*)take((size_t)MM * DD * 2);
  _Float16* qkv16  = (_Float16*)take((size_t)MM * 3 * DD * 2);
  _Float16* ao16   = (_Float16*)take((size_t)MM * DD * 2);
  float*    tf     = (float*)take((size_t)MM * DD * 4);
  float*    hid    = (float*)take((size_t)BB * DD * 4);
  // ff1 activations (32MB) alias the (dead-by-then) qkv16+ao16 region (24+8MB)
  _Float16* ff16 = qkv16;

  auto conv = [&](const float* s, _Float16* d, int n) {
    f32_to_f16_kernel<<<(n + 255) / 256, 256, 0, stream>>>(s, d, n);
  };
  conv(x, x16, MM * FF);
  conv(proj_w, wp16, DD * FF);
  conv(in_proj_w, wqkv16, LL * 3 * DD * DD);
  conv(out_proj_w, wo16, LL * DD * DD);
  conv(ff1_w, wf1_16, LL * DFFC * DD);
  conv(ff2_w, wf2_16, LL * DD * DFFC);

  dim3 blk(256);
  // embed: hf = x @ proj_w^T + proj_b
  gemm_wmma<false, true, false><<<dim3(DD / 64, MM / 256), blk, 0, stream>>>(
      x16, wp16, proj_b, hf, nullptr, MM, DD, FF);
  add_pe_kernel<<<(MM * DD) / 256, 256, 0, stream>>>(hf, h16);

  for (int l = 0; l < LL; ++l) {
    // qkv = h @ in_proj^T + b   (fp16 out, consumed by attention)
    gemm_wmma<false, false, true><<<dim3(3 * DD / 64, MM / 256), blk, 0, stream>>>(
        h16, wqkv16 + (size_t)l * 3 * DD * DD, in_proj_b + l * 3 * DD,
        nullptr, qkv16, MM, 3 * DD, DD);
    flash_attn_kernel<<<BB * HH * (TT / 64), dim3(128), 0, stream>>>(qkv16, ao16);
    // o = attn @ out_proj^T + b  (f32 out for residual)
    gemm_wmma<false, true, false><<<dim3(DD / 64, MM / 256), blk, 0, stream>>>(
        ao16, wo16 + (size_t)l * DD * DD, out_proj_b + l * DD,
        tf, nullptr, MM, DD, DD);
    ln_residual_kernel<<<MM, 256, 0, stream>>>(hf, tf, ln1_w + l * DD,
                                               ln1_b + l * DD, h16);
    // ff1 (relu, fp16 out)
    gemm_wmma<true, false, true><<<dim3(DFFC / 64, MM / 256), blk, 0, stream>>>(
        h16, wf1_16 + (size_t)l * DFFC * DD, ff1_b + l * DFFC,
        nullptr, ff16, MM, DFFC, DD);
    // ff2 (f32 out)
    gemm_wmma<false, true, false><<<dim3(DD / 64, MM / 256), blk, 0, stream>>>(
        ff16, wf2_16 + (size_t)l * DD * DFFC, ff2_b + l * DD,
        tf, nullptr, MM, DD, DFFC);
    ln_residual_kernel<<<MM, 256, 0, stream>>>(hf, tf, ln2_w + l * DD,
                                               ln2_b + l * DD, h16);
  }

  head1_kernel<<<BB, 256, 0, stream>>>(hf, mlp1_w, mlp1_b, hid);
  head2_kernel<<<1, 128, 0, stream>>>(hid, mlp2_w, mlp2_b, outp);
}
